// Graph_HiC_Likelihood_76871324664260
// MI455X (gfx1250) — matrix operand order repaired and checked by reading, not compiled, not run on hardware
//
#include <hip/hip_runtime.h>
#include <hip/hip_bf16.h>
#include <math.h>

// ---------------------------------------------------------------------------
// Graph_HiC_Likelihood forward for MI455X (gfx1250, wave32, WMMA).
//
// Pipeline:
//   1. gcn_norm for 7 edge attrs + learned edge weight (atomics for degree)
//   2. per-stage: GEMM (f16 WMMA 16x16x32, f32 acc, 32x64 supertile/wave) ->
//      self-loop init -> float4 atomic scatter -> relu + f16 convert
//   3. fused link MLP: 16 test edges per wave, layer0 K=1792 gathered from
//      x4 rows, 3 hidden layers through LDS, final 128->4 scalar.
// ---------------------------------------------------------------------------

typedef __attribute__((ext_vector_type(16))) _Float16 v16h;
typedef __attribute__((ext_vector_type(8)))  _Float16 v8h;
typedef __attribute__((ext_vector_type(8)))  float    v8f;

#define kN   20000
#define kE   200000
#define kET  40000
#define kH   128
#define kHC  896      /* 7*128 */
#define kKL  1792     /* 2*896 */

// ---------------------------------------------------------------------------
// helpers
// ---------------------------------------------------------------------------
static __device__ inline v16h make16(v8h lo, v8h hi) {
  v16h a;
#pragma unroll
  for (int j = 0; j < 8; ++j) { a[j] = lo[j]; a[j + 8] = hi[j]; }
  return a;
}

// Store a row of 8 C-fragments (16x128 f32) as relu->f16 into LDS [16][136].
static __device__ inline void c_store_lds(const v8f* c, const float* __restrict__ bias,
                                          _Float16* hb, int l) {
  const int n  = l & 15;
  const int mb = (l >> 4) << 3;
#pragma unroll
  for (int t = 0; t < 8; ++t) {
    float bv = bias[t * 16 + n];
#pragma unroll
    for (int r = 0; r < 8; ++r)
      hb[(mb + r) * 136 + t * 16 + n] = (_Float16)fmaxf(c[t][r] + bv, 0.f);
  }
}

// ---------------------------------------------------------------------------
// small elementwise kernels
// ---------------------------------------------------------------------------
__global__ void zero_f(float* __restrict__ p, int total) {
  int t = blockIdx.x * blockDim.x + threadIdx.x;
  if (t < total) p[t] = 0.f;
}

__global__ void cvt_x(const float* __restrict__ x, _Float16* __restrict__ xh, int total) {
  int t = blockIdx.x * blockDim.x + threadIdx.x;
  if (t >= total) return;
  int n = t >> 5, k = t & 31;                 // padded lda = 32, K>=8 zero
  xh[t] = (k < 8) ? (_Float16)x[n * 8 + k] : (_Float16)0.f;
}

// Pack weight (nea, Ksrc, Hh) -> f16 fragment-packed [K x Ncols], column-concat
// over nea, zero-padded for k >= Ksrc. Packed so a B fragment is lane-contiguous:
// element (k,n): tile=(k/32)*(Ncols/16)+(n/16); lane=(((k%32)/8)%2)*16+(n%16); j=(k%8)+((k%32)/16)*8.
__global__ void pack_w(const float* __restrict__ src, _Float16* __restrict__ dst,
                       int Ksrc, int K, int Hh, int Ncols) {
  int t = blockIdx.x * blockDim.x + threadIdx.x;
  if (t >= K * Ncols) return;
  int k = t / Ncols, n = t - k * Ncols;
  int i = n / Hh,   h = n - i * Hh;
  float v = (k < Ksrc) ? src[((size_t)i * Ksrc + k) * Hh + h] : 0.f;
  int kt = k >> 5, kr = k & 31, nt = n >> 4, nr = n & 15;
  int lane = (((kr >> 3) & 1) << 4) | nr, j = (kr & 7) | ((kr >> 4) << 3);
  dst[(((size_t)kt * (Ncols >> 4) + nt) << 9) + (lane << 4) + j] = (_Float16)v;
}

// ---------------------------------------------------------------------------
// gcn_norm pipeline
// ---------------------------------------------------------------------------
__global__ void deg_attr(const float* __restrict__ ea, const int* __restrict__ ei,
                         float* __restrict__ deg) {
  int e = blockIdx.x * blockDim.x + threadIdx.x;
  if (e >= kE) return;
  int c = ei[kE + e];
#pragma unroll
  for (int i = 0; i < 7; ++i) atomicAdd(&deg[i * kN + c], ea[e * 7 + i]);
}

// edge MLP: [7]->28->28->1 sigmoid, plus degree accumulation for the learned weight
__global__ void edge_mlp(const float* __restrict__ ea, const int* __restrict__ ei,
                         const float* __restrict__ ew1, const float* __restrict__ eb1,
                         const float* __restrict__ ew2, const float* __restrict__ eb2,
                         const float* __restrict__ ew3, const float* __restrict__ eb3,
                         float* __restrict__ ewv, float* __restrict__ deg7) {
  int e = blockIdx.x * blockDim.x + threadIdx.x;
  if (e >= kE) return;
  float a[7];
#pragma unroll
  for (int i = 0; i < 7; ++i) a[i] = ea[e * 7 + i];
  float h1[28];
#pragma unroll
  for (int o = 0; o < 28; ++o) {
    float s = eb1[o];
#pragma unroll
    for (int k = 0; k < 7; ++k) s += a[k] * ew1[k * 28 + o];
    h1[o] = fmaxf(s, 0.f);
  }
  float h2[28];
  for (int o = 0; o < 28; ++o) {
    float s = eb2[o];
#pragma unroll
    for (int k = 0; k < 28; ++k) s += h1[k] * ew2[k * 28 + o];
    h2[o] = fmaxf(s, 0.f);
  }
  float s = eb3[0];
#pragma unroll
  for (int k = 0; k < 28; ++k) s += h2[k] * ew3[k];
  float w = 1.f / (1.f + expf(-s));
  ewv[e] = w;
  atomicAdd(&deg7[ei[kE + e]], w);
}

__global__ void dis_k(float* __restrict__ deg, int total) {   // deg -> dis in place
  int t = blockIdx.x * blockDim.x + threadIdx.x;
  if (t < total) deg[t] = rsqrtf(deg[t] + 1.0f);              // +1 = self loop
}

__global__ void norms_k(const int* __restrict__ ei, const float* __restrict__ ea,
                        const float* __restrict__ ewv, const float* __restrict__ dis,
                        float* __restrict__ norms, float* __restrict__ normw) {
  int e = blockIdx.x * blockDim.x + threadIdx.x;
  if (e >= kE) return;
  int r = ei[e], c = ei[kE + e];
#pragma unroll
  for (int i = 0; i < 7; ++i)
    norms[(size_t)i * kE + e] = dis[i * kN + r] * ea[e * 7 + i] * dis[i * kN + c];
  normw[e] = dis[7 * kN + r] * ewv[e] * dis[7 * kN + c];
}

__global__ void selfn_k(const float* __restrict__ dis, float* __restrict__ selfn, int total) {
  int t = blockIdx.x * blockDim.x + threadIdx.x;
  if (t < total) selfn[t] = dis[t] * dis[t];                  // weight 1 self loop
}

// ---------------------------------------------------------------------------
// generic WMMA GEMM: C[M x Ncols] = A[M x K](f16,row) * Bp[K x Ncols](f16,packed)
// one wave per block; wave owns a 32-row x 64-col supertile (2 M-tiles x
// 4 N-tiles) so every B fragment feeds two WMMAs -> B fragments stay live in
// distinct registers and loads overlap the WMMA stream.
// ---------------------------------------------------------------------------
__global__ void gemm_f16_wmma(const _Float16* __restrict__ A,
                              const _Float16* __restrict__ Bp,
                              float* __restrict__ C, int K, int Ncols) {
  const int tilesN  = Ncols >> 4;
  const int groupsN = Ncols >> 6;
  const int mt = blockIdx.x / groupsN;          // 32-row supertile index
  const int ng = blockIdx.x - mt * groupsN;
  const int l  = threadIdx.x;
  const int hi = (l >> 4) << 3;
  v8f acc[2][4];
#pragma unroll
  for (int mi = 0; mi < 2; ++mi)
#pragma unroll
    for (int t = 0; t < 4; ++t) acc[mi][t] = {};
  const _Float16* arow0 = A + (size_t)((mt << 5) + (l & 15)) * K;
  const _Float16* arow1 = arow0 + (size_t)16 * K;
  for (int k0 = 0; k0 < K; k0 += 32) {
    __builtin_prefetch(arow0 + k0 + 32, 0, 1);   // speculative global_prefetch
    // ---- load phase: 2 A tiles + 4 B tiles, all distinct registers ----
    const _Float16* pa0 = arow0 + k0 + hi;
    const _Float16* pa1 = arow1 + k0 + hi;
    v16h a0 = make16(*(const v8h*)pa0, *(const v8h*)(pa0 + 16));
    v16h a1 = make16(*(const v8h*)pa1, *(const v8h*)(pa1 + 16));
    size_t tbase = ((size_t)(k0 >> 5) * tilesN + (ng << 2)) << 9;
    v16h bfr[4];
#pragma unroll
    for (int t = 0; t < 4; ++t) {
      const v8h* pb = (const v8h*)(Bp + tbase + ((size_t)t << 9) + (l << 4));
      bfr[t] = make16(pb[0], pb[1]);
    }
    // ---- compute phase: each B fragment consumed twice ----
#pragma unroll
    for (int t = 0; t < 4; ++t) {
      acc[0][t] = __builtin_amdgcn_wmma_f32_16x16x32_f16(false, a0, false, bfr[t],
                                                         (short)0, acc[0][t], false, false);
      acc[1][t] = __builtin_amdgcn_wmma_f32_16x16x32_f16(false, a1, false, bfr[t],
                                                         (short)0, acc[1][t], false, false);
    }
  }
  const int nb = (ng << 6) + (l & 15);
#pragma unroll
  for (int mi = 0; mi < 2; ++mi) {
    const int mrow = (mt << 5) + (mi << 4) + ((l >> 4) << 3);
#pragma unroll
    for (int t = 0; t < 4; ++t)
#pragma unroll
      for (int r = 0; r < 8; ++r)
        C[(size_t)(mrow + r) * Ncols + nb + t * 16] = acc[mi][t][r];
  }
}

// ---------------------------------------------------------------------------
// conv epilogue: self-loop init, edge scatter (atomics), relu+f16
// ---------------------------------------------------------------------------
__global__ void conv_init(float* __restrict__ out, const float* __restrict__ xw,
                          const float* __restrict__ bias, const float* __restrict__ selfn,
                          int Hout, int multi, int total) {
  int t = blockIdx.x * blockDim.x + threadIdx.x;
  if (t >= total) return;
  int n = t / Hout, j = t - n * Hout;
  int a = multi ? (j >> 7) : 0;
  out[t] = bias[j] + selfn[a * kN + n] * xw[t];
}

__global__ void conv_scatter(float* __restrict__ out, const float* __restrict__ xw,
                             const int* __restrict__ ei, const float* __restrict__ nrm,
                             int Hout, int multi, int total) {
  int t = blockIdx.x * blockDim.x + threadIdx.x;
  if (t >= total) return;
  int chunks = Hout >> 2;
  int e  = t / chunks;
  int jc = (t - e * chunks) << 2;
  int r = ei[e], c = ei[kE + e];
  float w = multi ? nrm[(size_t)(jc >> 7) * kE + e] : nrm[e];
  const float4 v = *(const float4*)(xw + (size_t)r * Hout + jc);
  float* po = out + (size_t)c * Hout + jc;
  atomicAdd(po + 0, w * v.x);
  atomicAdd(po + 1, w * v.y);
  atomicAdd(po + 2, w * v.z);
  atomicAdd(po + 3, w * v.w);
}

__global__ void relu_h(const float* __restrict__ in, _Float16* __restrict__ outh, int total) {
  int t = blockIdx.x * blockDim.x + threadIdx.x;
  if (t < total) outh[t] = (_Float16)fmaxf(in[t], 0.f);
}

// ---------------------------------------------------------------------------
// fused link MLP: one wave handles 16 test edges end-to-end.
//   layer0: [16 x 1792] x [1792 x 128] via gathered x4 rows
//   3 hidden layers 128->128 through LDS (padded 136-half rows, conflict free)
//   final 128->4 scalar, symmetrized 0.5*(t1 + t2[:, perm])
// ---------------------------------------------------------------------------
__global__ void link_mlp_wmma(const _Float16* __restrict__ x4, const int* __restrict__ eit,
                              const _Float16* __restrict__ lw0p, const float* __restrict__ lb0,
                              const _Float16* __restrict__ lwhp, const float* __restrict__ lbh,
                              const float* __restrict__ lw4, const float* __restrict__ lb4,
                              float* __restrict__ out, int dir) {
  __shared__ __align__(16) _Float16 hb[16 * 136];
  const int l  = threadIdx.x;
  const int m  = l & 15;
  const int hi = (l >> 4) << 3;
  const int et = (blockIdx.x << 4) + m;
  int s = eit[et], d = eit[kET + et];
  if (dir) { int tmp = s; s = d; d = tmp; }
  const _Float16* rs = x4 + (size_t)s * kHC;
  const _Float16* rd = x4 + (size_t)d * kHC;

  v8f acc[8];
#pragma unroll
  for (int t = 0; t < 8; ++t) acc[t] = {};
  for (int k0 = 0; k0 < kKL; k0 += 32) {
    int f = k0 + hi;                               // 896 % 32 == 0: chunk stays in one half
    const _Float16* base = (f < kHC) ? (rs + f) : (rd + (f - kHC));
    v16h a = make16(*(const v8h*)base, *(const v8h*)(base + 16));
    const _Float16* bb = lw0p + ((size_t)(k0 >> 5) << 12) + (l << 4);  // (k0/32)*8 tiles *512
    v16h bfr[8];
#pragma unroll
    for (int t = 0; t < 8; ++t) {
      const v8h* pb = (const v8h*)(bb + ((size_t)t << 9));
      bfr[t] = make16(pb[0], pb[1]);
    }
#pragma unroll
    for (int t = 0; t < 8; ++t)
      acc[t] = __builtin_amdgcn_wmma_f32_16x16x32_f16(false, a, false, bfr[t],
                                                      (short)0, acc[t], false, false);
  }
  c_store_lds(acc, lb0, hb, l);
  __syncthreads();

  for (int layer = 0; layer < 3; ++layer) {
    v8f h[8];
#pragma unroll
    for (int t = 0; t < 8; ++t) h[t] = {};
#pragma unroll
    for (int k0 = 0; k0 < kH; k0 += 32) {
      const _Float16* pa = hb + m * 136 + k0 + hi;
      v16h a = make16(*(const v8h*)pa, *(const v8h*)(pa + 16));
      const _Float16* bb = lwhp + (size_t)layer * kH * kH + ((size_t)(k0 >> 5) << 12) + (l << 4);
      v16h bfr[8];
#pragma unroll
      for (int t = 0; t < 8; ++t) {
        const v8h* pb = (const v8h*)(bb + ((size_t)t << 9));
        bfr[t] = make16(pb[0], pb[1]);
      }
#pragma unroll
      for (int t = 0; t < 8; ++t)
        h[t] = __builtin_amdgcn_wmma_f32_16x16x32_f16(false, a, false, bfr[t],
                                                      (short)0, h[t], false, false);
    }
    __syncthreads();
    c_store_lds(h, lbh + layer * kH, hb, l);
    __syncthreads();
  }

  const int perm[4] = {0, 2, 1, 3};
#pragma unroll
  for (int half = 0; half < 2; ++half) {
    int idx = l + (half << 5);
    int er = idx >> 2, o = idx & 3;
    float sum = lb4[o];
    for (int k = 0; k < kH; ++k) sum += (float)hb[er * 136 + k] * lw4[k * 4 + o];
    sum *= 0.5f;
    float* po = out + ((size_t)((blockIdx.x << 4) + er) << 2) + (dir ? perm[o] : o);
    if (dir) *po += sum; else *po = sum;
  }
}

// ---------------------------------------------------------------------------
// host: buffer layout + launch sequence (all on `stream`, graph-capture safe)
// ---------------------------------------------------------------------------
extern "C" void kernel_launch(void* const* d_in, const int* in_sizes, int n_in,
                              void* d_out, int out_size, void* d_ws, size_t ws_size,
                              hipStream_t stream) {
  (void)in_sizes; (void)n_in; (void)out_size; (void)ws_size;
  const float* x   = (const float*)d_in[0];
  const int*   ei  = (const int*)d_in[1];
  const float* ea  = (const float*)d_in[2];
  const int*   eit = (const int*)d_in[3];
  const float* W0  = (const float*)d_in[4];
  const float* b0  = (const float*)d_in[5];
  const float* W1  = (const float*)d_in[6];
  const float* b1  = (const float*)d_in[7];
  const float* W2  = (const float*)d_in[8];
  const float* b2  = (const float*)d_in[9];
  const float* W3  = (const float*)d_in[10];
  const float* b3  = (const float*)d_in[11];
  const float* ew1 = (const float*)d_in[12];
  const float* eb1 = (const float*)d_in[13];
  const float* ew2 = (const float*)d_in[14];
  const float* eb2 = (const float*)d_in[15];
  const float* ew3 = (const float*)d_in[16];
  const float* eb3 = (const float*)d_in[17];
  const float* lw0 = (const float*)d_in[18];
  const float* lb0 = (const float*)d_in[19];
  const float* lwh = (const float*)d_in[20];
  const float* lbh = (const float*)d_in[21];
  const float* lw4 = (const float*)d_in[22];
  const float* lb4 = (const float*)d_in[23];
  float* out = (float*)d_out;

  size_t off = 0;
  auto alloc = [&](size_t bytes) -> void* {
    off = (off + 255) & ~(size_t)255;
    void* p = (char*)d_ws + off;
    off += bytes;
    return p;
  };
  float*    deg   = (float*)alloc((size_t)8 * kN * 4);     // -> dis in place
  float*    ewv   = (float*)alloc((size_t)kE * 4);
  float*    norms = (float*)alloc((size_t)7 * kE * 4);
  float*    normw = (float*)alloc((size_t)kE * 4);
  float*    selfn = (float*)alloc((size_t)8 * kN * 4);
  float*    acc1  = (float*)alloc((size_t)kN * kHC * 4);   // GEMM output xw
  float*    acc2  = (float*)alloc((size_t)kN * kHC * 4);   // scatter accumulator
  _Float16* xh    = (_Float16*)alloc((size_t)kN * 32 * 2);
  _Float16* x1h   = (_Float16*)alloc((size_t)kN * kHC * 2);
  _Float16* x2h   = (_Float16*)alloc((size_t)kN * kH * 2);
  _Float16* x3h   = (_Float16*)alloc((size_t)kN * kHC * 2);
  _Float16* x4h   = (_Float16*)alloc((size_t)kN * kHC * 2);
  _Float16* W0p   = (_Float16*)alloc((size_t)32 * kHC * 2);
  _Float16* W1p   = (_Float16*)alloc((size_t)kHC * kH * 2);
  _Float16* W2p   = (_Float16*)alloc((size_t)kH * kHC * 2);
  _Float16* W3p   = (_Float16*)alloc((size_t)kHC * kHC * 2);
  _Float16* lw0p  = (_Float16*)alloc((size_t)kKL * kH * 2);
  _Float16* lwhp  = (_Float16*)alloc((size_t)3 * kH * kH * 2);

  auto g = [](int total, int bs) { return dim3((unsigned)((total + bs - 1) / bs)); };

  // ---- weight packing + input convert ----
  pack_w<<<g(32 * kHC, 256), 256, 0, stream>>>(W0, W0p, 8, 32, kH, kHC);
  pack_w<<<g(kHC * kH, 256), 256, 0, stream>>>(W1, W1p, kHC, kHC, kH, kH);
  pack_w<<<g(kH * kHC, 256), 256, 0, stream>>>(W2, W2p, kH, kH, kH, kHC);
  pack_w<<<g(kHC * kHC, 256), 256, 0, stream>>>(W3, W3p, kHC, kHC, kH, kHC);
  pack_w<<<g(kKL * kH, 256), 256, 0, stream>>>(lw0, lw0p, kKL, kKL, kH, kH);
  for (int t = 0; t < 3; ++t)
    pack_w<<<g(kH * kH, 256), 256, 0, stream>>>(lwh + t * kH * kH, lwhp + t * kH * kH,
                                                kH, kH, kH, kH);
  cvt_x<<<g(kN * 32, 256), 256, 0, stream>>>(x, xh, kN * 32);

  // ---- gcn_norm (7 attrs + learned edge weight) ----
  zero_f<<<g(8 * kN, 256), 256, 0, stream>>>(deg, 8 * kN);
  deg_attr<<<g(kE, 256), 256, 0, stream>>>(ea, ei, deg);
  edge_mlp<<<g(kE, 128), 128, 0, stream>>>(ea, ei, ew1, eb1, ew2, eb2, ew3, eb3,
                                           ewv, deg + 7 * kN);
  dis_k<<<g(8 * kN, 256), 256, 0, stream>>>(deg, 8 * kN);
  norms_k<<<g(kE, 256), 256, 0, stream>>>(ei, ea, ewv, deg, norms, normw);
  selfn_k<<<g(8 * kN, 256), 256, 0, stream>>>(deg, selfn, 8 * kN);

  // ---- conv stages: GEMM -> init -> scatter -> relu/f16 ----
  auto conv_stage = [&](const _Float16* Ain, int K, const _Float16* Bp, int Hout,
                        const float* bias, const float* selfPtr, const float* normPtr,
                        int multi, _Float16* outH) {
    gemm_f16_wmma<<<dim3((unsigned)((kN / 32) * (Hout / 64))), 32, 0, stream>>>(
        Ain, Bp, acc1, K, Hout);
    int tot = kN * Hout;
    conv_init<<<g(tot, 256), 256, 0, stream>>>(acc2, acc1, bias, selfPtr, Hout, multi, tot);
    int stot = kE * (Hout / 4);
    conv_scatter<<<g(stot, 256), 256, 0, stream>>>(acc2, acc1, ei, normPtr, Hout, multi, stot);
    relu_h<<<g(tot, 256), 256, 0, stream>>>(acc2, outH, tot);
  };
  conv_stage(xh,  32,  W0p, kHC, b0, selfn,          norms, 1, x1h);
  conv_stage(x1h, kHC, W1p, kH,  b1, selfn + 7 * kN, normw, 0, x2h);
  conv_stage(x2h, kH,  W2p, kHC, b2, selfn,          norms, 1, x3h);
  conv_stage(x3h, kHC, W3p, kHC, b3, selfn,          norms, 1, x4h);

  // ---- symmetrized link MLP ----
  link_mlp_wmma<<<dim3(kET / 16), 32, 0, stream>>>(x4h, eit, lw0p, lb0, lwhp, lbh,
                                                   lw4, lb4, out, 0);
  link_mlp_wmma<<<dim3(kET / 16), 32, 0, stream>>>(x4h, eit, lw0p, lb0, lwhp, lbh,
                                                   lw4, lb4, out, 1);
}
